// SpatialTemporalLayer_30227979829645
// MI455X (gfx1250) — matrix operand order repaired and checked by reading, not compiled, not run on hardware
//
#include <hip/hip_runtime.h>
#include <hip/hip_bf16.h>

// ---------------------------------------------------------------------------
// Problem constants (from reference)
// ---------------------------------------------------------------------------
#define B_   64
#define NS_  2
#define NF_  13
#define T_   24
#define D_   512
#define H_   8
#define DH_  64
#define DFF_ 2048
#define NTOK (B_ * NS_ * NF_ * T_)   // 39936 tokens

// ---------------------------------------------------------------------------
// CDNA5 (gfx1250, wave32) WMMA types
// ---------------------------------------------------------------------------
typedef __attribute__((ext_vector_type(16))) __bf16 v16bf;
typedef __attribute__((ext_vector_type(8)))  float  v8f;
typedef __attribute__((ext_vector_type(4)))  int    v4i;

// Async global->LDS DMA path (gfx1250). Guarded so both toolchains compile.
#if defined(__gfx1250__) && defined(__has_builtin)
#  if __has_builtin(__builtin_amdgcn_global_load_async_to_lds_b128)
#    define ASYNC_LDS 1
#  endif
#endif

#if defined(ASYNC_LDS)
typedef __attribute__((address_space(1))) v4i gv4i;   // global <4 x i32>
typedef __attribute__((address_space(3))) v4i lv4i;   // LDS    <4 x i32>
#endif

// f32 -> bf16 (round to nearest even)
__device__ __forceinline__ unsigned short f32_to_bf16(float x) {
    unsigned u = __float_as_uint(x);
    return (unsigned short)((u + 0x7fffu + ((u >> 16) & 1u)) >> 16);
}
__device__ __forceinline__ unsigned pack2_bf16(float x, float y) {
    return (unsigned)f32_to_bf16(x) | ((unsigned)f32_to_bf16(y) << 16);
}

// ---------------------------------------------------------------------------
// Bulk f32 -> bf16 conversion (vectorized, n must be a multiple of 4)
// ---------------------------------------------------------------------------
__global__ __launch_bounds__(256)
void f32_to_bf16_vec(const float* __restrict__ in, unsigned short* __restrict__ out, int n4)
{
    const int i = blockIdx.x * blockDim.x + threadIdx.x;
    if (i < n4) {
        const float4 v = ((const float4*)in)[i];
        ((uint2*)out)[i] = make_uint2(pack2_bf16(v.x, v.y), pack2_bf16(v.z, v.w));
    }
}

// ---------------------------------------------------------------------------
// GEMM: C[M,N] = A[M,K] * W[N,K]^T + bias[N]   (optional ReLU, optional bf16 out)
//   A, W are bf16 in global memory; staged raw into LDS (async DMA when
//   available); f32 accumulation via v_wmma_f32_16x16x32_bf16.
// Block tile: BM=128, BN=256, BK=32. 256 threads = 8 waves (wave32),
// wave grid 2(M) x 4(N): each wave owns a 64x64 tile = 4x4 WMMA accumulators.
// Requires M%128==0, N%256==0, K%32==0 (true for every call here).
// ---------------------------------------------------------------------------
#define TILE_M 128
#define TILE_N 256
#define TILE_K 32
#define LDS_STRIDE 40   // padded row stride (bf16 elems); 80 B, 16-B aligned

__device__ __forceinline__ void stage_chunk(const unsigned short* __restrict__ gsrc,
                                            unsigned short* __restrict__ ldst)
{
#if defined(ASYNC_LDS)
    __builtin_amdgcn_global_load_async_to_lds_b128(
        (gv4i*)gsrc, (lv4i*)ldst, 0, 0);
#else
    *(uint4*)ldst = *(const uint4*)gsrc;
#endif
}

template <bool RELU, bool OUT_BF16>
__global__ __launch_bounds__(256)
void gemm_bf16_wmma(const unsigned short* __restrict__ A,
                    const unsigned short* __restrict__ W,
                    const float* __restrict__ bias,
                    void* __restrict__ Cout,
                    int M, int N, int K)
{
    __shared__ __align__(16) unsigned short As[TILE_M * LDS_STRIDE];
    __shared__ __align__(16) unsigned short Bs[TILE_N * LDS_STRIDE];

    const int tid    = threadIdx.x;
    const int lane   = tid & 31;
    const int wave   = tid >> 5;
    const int wm     = wave >> 2;        // 0..1  -> 64-row slab
    const int wn     = wave & 3;         // 0..3  -> 64-col slab
    const int blockM = blockIdx.y * TILE_M;
    const int blockN = blockIdx.x * TILE_N;

    const int half = (lane < 16) ? 0 : 1;   // K-half held by this lane
    const int rsel = lane & 15;             // row (A) / col (B) within 16

    v8f acc[4][4];
    #pragma unroll
    for (int mt = 0; mt < 4; ++mt)
        #pragma unroll
        for (int nt = 0; nt < 4; ++nt)
            #pragma unroll
            for (int e = 0; e < 8; ++e)
                acc[mt][nt][e] = 0.0f;

    union Frag { v16bf v; uint4 q[2]; };

    for (int k0 = 0; k0 < K; k0 += TILE_K) {
        // ---- stage A (128x32) and B (256x32) bf16 tiles into LDS ----------
        // 16-byte chunks: A has 512 (2/thread), B has 1024 (4/thread).
        #pragma unroll
        for (int i = 0; i < 2; ++i) {
            const int c   = tid + 256 * i;
            const int row = c >> 2;
            const int seg = (c & 3) << 3;          // 0,8,16,24 (bf16 elems)
            stage_chunk(A + (size_t)(blockM + row) * K + k0 + seg,
                        &As[row * LDS_STRIDE + seg]);
        }
        #pragma unroll
        for (int i = 0; i < 4; ++i) {
            const int c   = tid + 256 * i;
            const int row = c >> 2;
            const int seg = (c & 3) << 3;
            stage_chunk(W + (size_t)(blockN + row) * K + k0 + seg,
                        &Bs[row * LDS_STRIDE + seg]);
        }
#if defined(ASYNC_LDS)
#  if __has_builtin(__builtin_amdgcn_s_wait_asynccnt)
        __builtin_amdgcn_s_wait_asynccnt(0);
#  else
        asm volatile("s_wait_asynccnt 0x0" ::: "memory");
#  endif
#else
        if (k0 + TILE_K < K) {
            __builtin_prefetch(A + (size_t)(blockM + (tid >> 1)) * K + k0 + TILE_K, 0, 1);
            __builtin_prefetch(W + (size_t)(blockN + (tid >> 1)) * K + k0 + TILE_K, 0, 1);
        }
#endif
        __syncthreads();

        // ---- fragments per ISA 16-bit layouts -----------------------------
        // A (16x32): lanes 0-15 row M=l, K {0..7,16..23}; lanes 16-31 K {8..15,24..31}
        Frag a[4];
        #pragma unroll
        for (int mt = 0; mt < 4; ++mt) {
            const unsigned short* p = &As[(wm * 64 + mt * 16 + rsel) * LDS_STRIDE + half * 8];
            a[mt].q[0] = *(const uint4*)(p);
            a[mt].q[1] = *(const uint4*)(p + 16);
        }
        // B (32x16): lanes 0-15 col N=l, K {0..7,16..23}; lanes 16-31 K {8..15,24..31}
        Frag b[4];
        #pragma unroll
        for (int nt = 0; nt < 4; ++nt) {
            const unsigned short* p = &Bs[(wn * 64 + nt * 16 + rsel) * LDS_STRIDE + half * 8];
            b[nt].q[0] = *(const uint4*)(p);
            b[nt].q[1] = *(const uint4*)(p + 16);
        }

        // ---- 16 WMMAs per K-step per wave ---------------------------------
        #pragma unroll
        for (int mt = 0; mt < 4; ++mt)
            #pragma unroll
            for (int nt = 0; nt < 4; ++nt)
                acc[mt][nt] = __builtin_amdgcn_wmma_f32_16x16x32_bf16(
                    false, a[mt].v, false, b[nt].v,
                    (short)0, acc[mt][nt], false, false);

        __syncthreads();
    }

    // ---- epilogue: bias (+ReLU), store f32 or bf16 ------------------------
    // C/D layout: VGPR r: lanes 0-15 -> (M=r, N=lane), lanes 16-31 -> (M=r+8, N=lane-16)
    #pragma unroll
    for (int nt = 0; nt < 4; ++nt) {
        const int col = blockN + wn * 64 + nt * 16 + rsel;
        const float bv = bias[col];
        #pragma unroll
        for (int mt = 0; mt < 4; ++mt) {
            const int rowbase = blockM + wm * 64 + mt * 16 + (half ? 8 : 0);
            #pragma unroll
            for (int r = 0; r < 8; ++r) {
                float v = acc[mt][nt][r] + bv;
                if (RELU) v = fmaxf(v, 0.0f);
                if (OUT_BF16)
                    ((unsigned short*)Cout)[(size_t)(rowbase + r) * N + col] = f32_to_bf16(v);
                else
                    ((float*)Cout)[(size_t)(rowbase + r) * N + col] = v;
            }
        }
    }
}

// ---------------------------------------------------------------------------
// Tiny-sequence multi-head attention (seq length S = 2 or 13).
// One wave per (batch, outer, t, head). Lane holds dims {lane, lane+32} of DH=64.
// qkv (f32) rows: [row][0:512]=Q, [512:1024]=K, [1024:1536]=V, head h at h*64.
// ctx is written in bf16 (it only feeds the out-projection GEMM).
// token row(m) = ((b*26 + m*strideM + outer*strideO)*24 + t)
// ---------------------------------------------------------------------------
template <int S>
__global__ __launch_bounds__(256)
void attn_smallseq(const float* __restrict__ qkv, unsigned short* __restrict__ ctx,
                   int strideM, int strideO, int nOuter, int totalWarps)
{
    const int warp = (int)(blockIdx.x * blockDim.x + threadIdx.x) >> 5;
    if (warp >= totalWarps) return;
    const int lane = threadIdx.x & 31;

    const int h = warp & 7;
    int g = warp >> 3;
    const int t = g % T_;  g /= T_;
    const int outer = g % nOuter;
    const int b = g / nOuter;

    int   rows[S];
    float q0[S], q1[S], k0v[S], k1v[S], v0[S], v1[S];
    #pragma unroll
    for (int m = 0; m < S; ++m) {
        const int row = ((b * (NS_ * NF_) + m * strideM + outer * strideO) * T_ + t);
        rows[m] = row;
        const float* p = qkv + (size_t)row * (3 * D_) + h * DH_ + lane;
        q0[m]  = p[0];        q1[m]  = p[32];
        k0v[m] = p[D_];       k1v[m] = p[D_ + 32];
        v0[m]  = p[2 * D_];   v1[m]  = p[2 * D_ + 32];
    }

    #pragma unroll
    for (int q = 0; q < S; ++q) {
        float s[S];
        #pragma unroll
        for (int k = 0; k < S; ++k) {
            float d = q0[q] * k0v[k] + q1[q] * k1v[k];
            #pragma unroll
            for (int off = 16; off > 0; off >>= 1)
                d += __shfl_xor(d, off);
            s[k] = d * 0.125f;    // 1/sqrt(DH=64)
        }
        float mx = s[0];
        #pragma unroll
        for (int k = 1; k < S; ++k) mx = fmaxf(mx, s[k]);
        float sum = 0.0f;
        #pragma unroll
        for (int k = 0; k < S; ++k) { s[k] = __expf(s[k] - mx); sum += s[k]; }
        const float inv = 1.0f / sum;
        float o0 = 0.0f, o1 = 0.0f;
        #pragma unroll
        for (int k = 0; k < S; ++k) {
            const float w = s[k] * inv;
            o0 += w * v0[k];
            o1 += w * v1[k];
        }
        unsigned short* po = ctx + (size_t)rows[q] * D_ + h * DH_ + lane;
        po[0]  = f32_to_bf16(o0);
        po[32] = f32_to_bf16(o1);
    }
}

// ---------------------------------------------------------------------------
// out = LayerNorm(a + r) * gamma + beta       (row width D_=512)
// Optionally also emits a bf16 copy (when the result feeds a GEMM).
// One block (256 threads) per row; each thread owns 2 elements.
// ---------------------------------------------------------------------------
template <bool WB16>
__global__ __launch_bounds__(256)
void ln_residual(const float* __restrict__ a, const float* __restrict__ r,
                 float* __restrict__ o, unsigned short* __restrict__ ob,
                 const float* __restrict__ gamma, const float* __restrict__ beta)
{
    const int row = blockIdx.x;
    const int tid = threadIdx.x;
    const float* pa = a + (size_t)row * D_;
    const float* pr = r + (size_t)row * D_;

    float v0 = pa[tid]       + pr[tid];
    float v1 = pa[tid + 256] + pr[tid + 256];

    __shared__ float red[8];
    __shared__ float stat[2];

    float s = v0 + v1;
    #pragma unroll
    for (int off = 16; off > 0; off >>= 1) s += __shfl_xor(s, off);
    if ((tid & 31) == 0) red[tid >> 5] = s;
    __syncthreads();
    if (tid == 0) {
        float tsum = 0.0f;
        #pragma unroll
        for (int i = 0; i < 8; ++i) tsum += red[i];
        stat[0] = tsum * (1.0f / (float)D_);
    }
    __syncthreads();
    const float mu = stat[0];
    const float d0 = v0 - mu, d1 = v1 - mu;

    float ss = d0 * d0 + d1 * d1;
    #pragma unroll
    for (int off = 16; off > 0; off >>= 1) ss += __shfl_xor(ss, off);
    if ((tid & 31) == 0) red[tid >> 5] = ss;
    __syncthreads();
    if (tid == 0) {
        float tsum = 0.0f;
        #pragma unroll
        for (int i = 0; i < 8; ++i) tsum += red[i];
        stat[1] = rsqrtf(tsum * (1.0f / (float)D_) + 1e-5f);
    }
    __syncthreads();
    const float inv = stat[1];

    const float y0 = d0 * inv * gamma[tid]       + beta[tid];
    const float y1 = d1 * inv * gamma[tid + 256] + beta[tid + 256];
    o[(size_t)row * D_ + tid]       = y0;
    o[(size_t)row * D_ + tid + 256] = y1;
    if (WB16) {
        ob[(size_t)row * D_ + tid]       = f32_to_bf16(y0);
        ob[(size_t)row * D_ + tid + 256] = f32_to_bf16(y1);
    }
}

// ---------------------------------------------------------------------------
// Orchestration
// ---------------------------------------------------------------------------
extern "C" void kernel_launch(void* const* d_in, const int* in_sizes, int n_in,
                              void* d_out, int out_size, void* d_ws, size_t ws_size,
                              hipStream_t stream)
{
    const float* x       = (const float*)d_in[0];
    const float* w_in_s  = (const float*)d_in[1];
    const float* b_in_s  = (const float*)d_in[2];
    const float* w_out_s = (const float*)d_in[3];
    const float* b_out_s = (const float*)d_in[4];
    const float* w_in_f  = (const float*)d_in[5];
    const float* b_in_f  = (const float*)d_in[6];
    const float* w_out_f = (const float*)d_in[7];
    const float* b_out_f = (const float*)d_in[8];
    const float* w1      = (const float*)d_in[9];
    const float* b1      = (const float*)d_in[10];
    const float* w2      = (const float*)d_in[11];
    const float* b2      = (const float*)d_in[12];
    const float* g1      = (const float*)d_in[13];
    const float* be1     = (const float*)d_in[14];
    const float* g2      = (const float*)d_in[15];
    const float* be2     = (const float*)d_in[16];
    const float* g3      = (const float*)d_in[17];
    const float* be3     = (const float*)d_in[18];
    float* out = (float*)d_out;

    // ---- workspace layout (256-B aligned slabs) ----
    char*  wsp = (char*)d_ws;
    size_t off = 0;
    auto alloc = [&](size_t bytes) -> void* {
        void* p = wsp + off;
        off += (bytes + 255) & ~(size_t)255;
        return p;
    };
    float*          qkv  = (float*)alloc((size_t)NTOK * 3 * D_ * 4);  // also reused as bf16 FFN hidden
    unsigned short* hidb = (unsigned short*)qkv;                       // NTOK*DFF_*2 < qkv bytes
    unsigned short* ctxb = (unsigned short*)alloc((size_t)NTOK * D_ * 2);
    float*          proj = (float*)alloc((size_t)NTOK * D_ * 4);
    float*          x1   = (float*)alloc((size_t)NTOK * D_ * 4);
    float*          x2   = (float*)alloc((size_t)NTOK * D_ * 4);
    unsigned short* xb   = (unsigned short*)alloc((size_t)NTOK * D_ * 2);
    unsigned short* x1b  = (unsigned short*)alloc((size_t)NTOK * D_ * 2);
    unsigned short* x2b  = (unsigned short*)alloc((size_t)NTOK * D_ * 2);
    unsigned short* wisb = (unsigned short*)alloc((size_t)3 * D_ * D_ * 2);
    unsigned short* wosb = (unsigned short*)alloc((size_t)D_ * D_ * 2);
    unsigned short* wifb = (unsigned short*)alloc((size_t)3 * D_ * D_ * 2);
    unsigned short* wofb = (unsigned short*)alloc((size_t)D_ * D_ * 2);
    unsigned short* w1b  = (unsigned short*)alloc((size_t)DFF_ * D_ * 2);
    unsigned short* w2b  = (unsigned short*)alloc((size_t)D_ * DFF_ * 2);

    const dim3 blk(256);
    auto conv = [&](const float* src, unsigned short* dst, size_t n) {
        const int n4 = (int)(n / 4);
        f32_to_bf16_vec<<<dim3((n4 + 255) / 256), blk, 0, stream>>>(src, dst, n4);
    };

    // ---- one-time f32 -> bf16 conversions (weights + input) ----
    conv(x,       xb,   (size_t)NTOK * D_);
    conv(w_in_s,  wisb, (size_t)3 * D_ * D_);
    conv(w_out_s, wosb, (size_t)D_ * D_);
    conv(w_in_f,  wifb, (size_t)3 * D_ * D_);
    conv(w_out_f, wofb, (size_t)D_ * D_);
    conv(w1,      w1b,  (size_t)DFF_ * D_);
    conv(w2,      w2b,  (size_t)D_ * DFF_);

    const int  mGrid = NTOK / TILE_M;              // 312
    const dim3 gQKV(3 * D_ / TILE_N, mGrid);       // (6, 312)
    const dim3 gOUT(D_ / TILE_N,     mGrid);       // (2, 312)
    const dim3 gFF1(DFF_ / TILE_N,   mGrid);       // (8, 312)

    // ---- station attention: attend over NS (S=2) ----
    gemm_bf16_wmma<false, false><<<gQKV, blk, 0, stream>>>(xb, wisb, b_in_s, qkv, NTOK, 3 * D_, D_);
    {
        const int totalWarps = B_ * NF_ * T_ * H_;              // 159744
        attn_smallseq<NS_><<<dim3((totalWarps * 32 + 255) / 256), blk, 0, stream>>>(
            qkv, ctxb, /*strideM=*/NF_, /*strideO=*/1, /*nOuter=*/NF_, totalWarps);
    }
    gemm_bf16_wmma<false, false><<<gOUT, blk, 0, stream>>>(ctxb, wosb, b_out_s, proj, NTOK, D_, D_);
    ln_residual<true><<<dim3(NTOK), blk, 0, stream>>>(x, proj, x1, x1b, g1, be1);

    // ---- feature attention: attend over NF (S=13) ----
    gemm_bf16_wmma<false, false><<<gQKV, blk, 0, stream>>>(x1b, wifb, b_in_f, qkv, NTOK, 3 * D_, D_);
    {
        const int totalWarps = B_ * NS_ * T_ * H_;              // 24576
        attn_smallseq<NF_><<<dim3((totalWarps * 32 + 255) / 256), blk, 0, stream>>>(
            qkv, ctxb, /*strideM=*/1, /*strideO=*/NF_, /*nOuter=*/NS_, totalWarps);
    }
    gemm_bf16_wmma<false, false><<<gOUT, blk, 0, stream>>>(ctxb, wofb, b_out_f, proj, NTOK, D_, D_);
    ln_residual<true><<<dim3(NTOK), blk, 0, stream>>>(x1, proj, x2, x2b, g2, be2);

    // ---- feed-forward ----
    gemm_bf16_wmma<true,  true ><<<gFF1, blk, 0, stream>>>(x2b,  w1b, b1, hidb, NTOK, DFF_, D_);
    gemm_bf16_wmma<false, false><<<gOUT, blk, 0, stream>>>(hidb, w2b, b2, proj, NTOK, D_,   DFF_);
    ln_residual<false><<<dim3(NTOK), blk, 0, stream>>>(x2, proj, out, (unsigned short*)nullptr, g3, be3);
}
